// ParallelxLSTMCell_66760971649453
// MI455X (gfx1250) — compile-verified, matrix operand order checked by reading, not run on hardware
//
#include <hip/hip_runtime.h>
#include <math.h>

// ---------------------------------------------------------------------------
// xLSTM parallel step for MI455X (gfx1250), reformulated as causal linear
// attention so the (B,S,D,D) cumsum never touches HBM. All GEMMs use
// v_wmma_f32_16x16x32_bf16 (wave32 WMMA), f32 accumulation, 32x32 register
// tiles per wave for 2x operand reuse.
// B=4, S=1024, IN=512, D=128.
// ---------------------------------------------------------------------------

#define B_  4
#define S_  1024
#define IN_ 512
#define D_  128

typedef __attribute__((ext_vector_type(16))) __bf16        v16bf;
typedef __attribute__((ext_vector_type(8)))  float         v8f;
typedef __attribute__((ext_vector_type(8)))  unsigned short v8us;

union BFU { unsigned short u; __bf16 b; };

static __device__ inline unsigned short f2bf(float f) {
    union { float f; unsigned int u; } c; c.f = f;
    unsigned int r = c.u + 0x7fffu + ((c.u >> 16) & 1u);   // RNE
    return (unsigned short)(r >> 16);
}
static __device__ inline float bf2f(unsigned short s) {
    union { unsigned int u; float f; } c; c.u = ((unsigned int)s) << 16;
    return c.f;
}
static __device__ inline __bf16 bfbits(unsigned short s) { BFU x; x.u = s; return x.b; }

// Load a 16x32 bf16 A/B fragment for lane (mn, half): the lane needs elements
// k = half*8 + 0..7 and 16 + half*8 + 0..7 -> two contiguous 16B loads.
static __device__ inline v16bf load_frag(const unsigned short* p) {
    v8us lo = *(const v8us*)p;
    v8us hi = *(const v8us*)(p + 16);
    v16bf f;
#pragma unroll
    for (int j = 0; j < 8; j++) { f[j] = bfbits(lo[j]); f[j + 8] = bfbits(hi[j]); }
    return f;
}

static __device__ inline v8f wmma_bf16(v16bf a, v16bf b, v8f c) {
    return __builtin_amdgcn_wmma_f32_16x16x32_bf16(false, a, false, b, (short)0, c,
                                                   false, false);
}

// ---------------------------------------------------------------------------
// 1) LayerNorm + i_tilde/f_tilde (1-dim projections) fused.
// ---------------------------------------------------------------------------
__global__ void ln_kernel(const float* __restrict__ x,
                          const float* __restrict__ g, const float* __restrict__ beta,
                          const float* __restrict__ Wi, const float* __restrict__ bi,
                          const float* __restrict__ Wf, const float* __restrict__ bfv,
                          unsigned short* __restrict__ xn,
                          float* __restrict__ itld, float* __restrict__ ftld)
{
    int row = blockIdx.x;
    int tid = threadIdx.x;
    const float* xr = x + (size_t)row * IN_;
    float v0 = xr[tid], v1 = xr[tid + 256];
    float s = v0 + v1, s2 = v0 * v0 + v1 * v1;
    __shared__ float sb[16];
#pragma unroll
    for (int o = 16; o > 0; o >>= 1) { s += __shfl_xor(s, o, 32); s2 += __shfl_xor(s2, o, 32); }
    int w = tid >> 5;
    if ((tid & 31) == 0) { sb[w] = s; sb[8 + w] = s2; }
    __syncthreads();
    float ts = 0.f, ts2 = 0.f;
#pragma unroll
    for (int i = 0; i < 8; i++) { ts += sb[i]; ts2 += sb[8 + i]; }
    float mu  = ts / (float)IN_;
    float var = ts2 / (float)IN_ - mu * mu;
    float inv = rsqrtf(var + 1e-5f);
    float xn0 = (v0 - mu) * inv * g[tid]       + beta[tid];
    float xn1 = (v1 - mu) * inv * g[tid + 256] + beta[tid + 256];
    xn[(size_t)row * IN_ + tid]       = f2bf(xn0);
    xn[(size_t)row * IN_ + tid + 256] = f2bf(xn1);
    float di = xn0 * Wi[tid] + xn1 * Wi[tid + 256];
    float df = xn0 * Wf[tid] + xn1 * Wf[tid + 256];
#pragma unroll
    for (int o = 16; o > 0; o >>= 1) { di += __shfl_xor(di, o, 32); df += __shfl_xor(df, o, 32); }
    __syncthreads();
    if ((tid & 31) == 0) { sb[w] = di; sb[8 + w] = df; }
    __syncthreads();
    if (tid == 0) {
        float a = 0.f, c = 0.f;
#pragma unroll
        for (int i = 0; i < 8; i++) { a += sb[i]; c += sb[8 + i]; }
        itld[row] = a + bi[0];
        ftld[row] = c + bfv[0];
    }
}

// ---------------------------------------------------------------------------
// 2) f32 -> bf16 conversion (weights, C_prev)
// ---------------------------------------------------------------------------
__global__ void cvt_bf(const float* __restrict__ s, unsigned short* __restrict__ d, int n)
{
    int i = blockIdx.x * blockDim.x + threadIdx.x;
    if (i < n) d[i] = f2bf(s[i]);
}

// ---------------------------------------------------------------------------
// 3) Sequential gate scan per batch: m_t, i, f, cumprod(f). Trivial cost.
// ---------------------------------------------------------------------------
__global__ void scan_kernel(const float* __restrict__ itld, const float* __restrict__ ftld,
                            const float* __restrict__ m_prev,
                            float* __restrict__ ia, float* __restrict__ fc,
                            float* __restrict__ out_m)
{
    if (threadIdx.x != 0) return;
    int b = blockIdx.x;
    float mp = m_prev[b];
    float fcv = 1.f, mlast = 0.f;
    for (int s = 0; s < S_; s++) {
        float itv = itld[b * S_ + s], ftv = ftld[b * S_ + s];
        float mt = fmaxf(ftv + mp, itv);
        float iv = expf(itv - mt);
        float fv = expf(ftv + mp - mt);
        fcv *= fv;
        ia[b * S_ + s] = iv;
        fc[b * S_ + s] = fcv;
        mlast = mt;
    }
    out_m[b] = mlast;
}

// ---------------------------------------------------------------------------
// 4) Projection GEMM, 32x32 register tile per wave:
//    2 A-frags x 2 B-frags -> 4 WMMAs per 4 fragment loads.
// ---------------------------------------------------------------------------
__global__ void gemm32(const unsigned short* __restrict__ A,
                       const unsigned short* __restrict__ W,
                       const float* __restrict__ bias,
                       float* __restrict__ C,
                       int M, int N, int K, float scale)
{
    int lane = threadIdx.x & 31;
    int half = lane >> 4, mn = lane & 15;
    int m0 = blockIdx.x * 32, n0 = blockIdx.y * 32;
    const unsigned short* a0p = A + (size_t)(m0 + mn) * K + half * 8;
    const unsigned short* a1p = A + (size_t)(m0 + 16 + mn) * K + half * 8;
    const unsigned short* b0p = W + (size_t)(n0 + mn) * K + half * 8;
    const unsigned short* b1p = W + (size_t)(n0 + 16 + mn) * K + half * 8;
    v8f acc00 = {}, acc01 = {}, acc10 = {}, acc11 = {};
    for (int kb = 0; kb < K; kb += 32) {
        __builtin_prefetch(a0p + kb + 128, 0, 0);   // speculative global_prefetch
        __builtin_prefetch(b0p + kb + 128, 0, 0);
        v16bf a0 = load_frag(a0p + kb);
        v16bf a1 = load_frag(a1p + kb);
        v16bf b0 = load_frag(b0p + kb);
        v16bf b1 = load_frag(b1p + kb);
        acc00 = wmma_bf16(a0, b0, acc00);
        acc01 = wmma_bf16(a0, b1, acc01);
        acc10 = wmma_bf16(a1, b0, acc10);
        acc11 = wmma_bf16(a1, b1, acc11);
    }
#pragma unroll
    for (int r = 0; r < 8; r++) {
        int row0 = m0 + r + 8 * half;
        int row1 = row0 + 16;
        int col0 = n0 + mn, col1 = col0 + 16;
        C[(size_t)row0 * N + col0] = (acc00[r] + bias[col0]) * scale;
        C[(size_t)row0 * N + col1] = (acc01[r] + bias[col1]) * scale;
        C[(size_t)row1 * N + col0] = (acc10[r] + bias[col0]) * scale;
        C[(size_t)row1 * N + col1] = (acc11[r] + bias[col1]) * scale;
    }
}

// ---------------------------------------------------------------------------
// 5) Pack: q->bf16, kk = i*k -> bf16 (row-major + transposed), v -> vT bf16.
// ---------------------------------------------------------------------------
__global__ void pack_kernel(const float* __restrict__ qf, const float* __restrict__ kf,
                            const float* __restrict__ vf, const float* __restrict__ ia,
                            unsigned short* __restrict__ qbf, unsigned short* __restrict__ kkbf,
                            unsigned short* __restrict__ kkT, unsigned short* __restrict__ vT)
{
    int row = blockIdx.x;          // b*S + s
    int d   = threadIdx.x;         // 0..127
    int b = row >> 10, s = row & (S_ - 1);
    float iv = ia[row];
    size_t idx  = (size_t)row * D_ + d;
    size_t tidx = ((size_t)(b * D_ + d)) * S_ + s;
    qbf[idx] = f2bf(qf[idx]);
    float kkv = kf[idx] * iv;
    kkbf[idx] = f2bf(kkv);
    kkT[tidx] = f2bf(kkv);
    vT[tidx]  = f2bf(vf[idx]);
}

// ---------------------------------------------------------------------------
// 6) Causal linear attention: one wave per (batch, 32-query tile).
//    acc starts as f_cum * (Q C_prev^T) (WMMA accumulation is linear), then
//    accumulates mask(Q kk^T) @ V. Every kk / V fragment is reused by the two
//    16-row query subtiles.
// ---------------------------------------------------------------------------
__global__ void attn_kernel(const unsigned short* __restrict__ qbf,
                            const unsigned short* __restrict__ kkbf,
                            const unsigned short* __restrict__ vT,
                            const unsigned short* __restrict__ cpbf,
                            const float* __restrict__ opre,
                            const float* __restrict__ fc,
                            const float* __restrict__ nprev,
                            float* __restrict__ outh)
{
    int b  = blockIdx.y;
    int m0 = blockIdx.x * 32;
    int lane = threadIdx.x & 31;
    int half = lane >> 4, mn = lane & 15;
    const unsigned short* qb  = qbf  + (size_t)b * S_ * D_;
    const unsigned short* kb  = kkbf + (size_t)b * S_ * D_;
    const unsigned short* vtb = vT   + (size_t)b * D_ * S_;
    const unsigned short* cpb = cpbf + (size_t)b * D_ * D_;

    __shared__ unsigned short lsc[32 * 32];

    // Q fragments: 2 row-subtiles x 4 feature chunks, register-resident.
    v16bf Aq[2][4];
#pragma unroll
    for (int u = 0; u < 2; u++)
#pragma unroll
        for (int c = 0; c < 4; c++)
            Aq[u][c] = load_frag(qb + (size_t)(m0 + u * 16 + mn) * D_ + c * 32 + half * 8);

    // per-row f_cum (uniform across the 16 lanes of each half-group)
    float fcr[2][8];
#pragma unroll
    for (int u = 0; u < 2; u++)
#pragma unroll
        for (int r = 0; r < 8; r++)
            fcr[u][r] = fc[b * S_ + m0 + u * 16 + r + 8 * half];

    // acc = Q @ C_prev^T, then scaled in place by f_cum.
    v8f zero = {};
    v8f acc[2][8];
#pragma unroll
    for (int u = 0; u < 2; u++)
#pragma unroll
        for (int dt = 0; dt < 8; dt++) acc[u][dt] = zero;
#pragma unroll
    for (int dt = 0; dt < 8; dt++)
#pragma unroll
        for (int c = 0; c < 4; c++) {
            v16bf Bc = load_frag(cpb + (size_t)(dt * 16 + mn) * D_ + c * 32 + half * 8);
#pragma unroll
            for (int u = 0; u < 2; u++) acc[u][dt] = wmma_bf16(Aq[u][c], Bc, acc[u][dt]);
        }
#pragma unroll
    for (int u = 0; u < 2; u++)
#pragma unroll
        for (int dt = 0; dt < 8; dt++)
#pragma unroll
            for (int r = 0; r < 8; r++) acc[u][dt][r] *= fcr[u][r];

    float qn[2][8];
#pragma unroll
    for (int u = 0; u < 2; u++)
#pragma unroll
        for (int r = 0; r < 8; r++) qn[u][r] = 0.f;

    int nch = m0 / 32 + 1;                    // key chunks of 32 covering keys 0..m0+31
    for (int ch = 0; ch < nch; ch++) {
#pragma unroll
        for (int ht = 0; ht < 2; ht++) {
            int kt0 = ch * 32 + ht * 16;
            v8f Sa0 = zero, Sa1 = zero;
#pragma unroll
            for (int c = 0; c < 4; c++) {
                v16bf Bk = load_frag(kb + (size_t)(kt0 + mn) * D_ + c * 32 + half * 8);
                Sa0 = wmma_bf16(Aq[0][c], Bk, Sa0);
                Sa1 = wmma_bf16(Aq[1][c], Bk, Sa1);
            }
            int key = kt0 + mn;
#pragma unroll
            for (int r = 0; r < 8; r++) {
                int m = r + 8 * half;
                float v0 = (key <= m0 + m)      ? Sa0[r] : 0.f;
                float v1 = (key <= m0 + 16 + m) ? Sa1[r] : 0.f;
                qn[0][r] += v0;
                qn[1][r] += v1;
                lsc[m * 32 + ht * 16 + mn]        = f2bf(v0);
                lsc[(16 + m) * 32 + ht * 16 + mn] = f2bf(v1);
            }
        }
        __syncthreads();                       // single-wave block: ordering fence
        v16bf Asc0 = load_frag(lsc + mn * 32 + half * 8);
        v16bf Asc1 = load_frag(lsc + (16 + mn) * 32 + half * 8);
#pragma unroll
        for (int dt = 0; dt < 8; dt++) {
            v16bf Bv = load_frag(vtb + (size_t)(dt * 16 + mn) * S_ + ch * 32 + half * 8);
            acc[0][dt] = wmma_bf16(Asc0, Bv, acc[0][dt]);
            acc[1][dt] = wmma_bf16(Asc1, Bv, acc[1][dt]);
        }
        __syncthreads();
    }

    // denominators: rowsum of scores + f_cum * (n_prev . q)
    float dnm[2][8];
#pragma unroll
    for (int u = 0; u < 2; u++)
#pragma unroll
        for (int r = 0; r < 8; r++) {
            float v = qn[u][r];
            v += __shfl_xor(v, 1, 32); v += __shfl_xor(v, 2, 32);
            v += __shfl_xor(v, 4, 32); v += __shfl_xor(v, 8, 32);
            int row = m0 + u * 16 + r + 8 * half;
            float p = 0.f;
#pragma unroll
            for (int e = 0; e < 8; e++) {
                int ei = mn * 8 + e;
                p += nprev[b * D_ + ei] * bf2f(qb[(size_t)row * D_ + ei]);
            }
            p += __shfl_xor(p, 1, 32); p += __shfl_xor(p, 2, 32);
            p += __shfl_xor(p, 4, 32); p += __shfl_xor(p, 8, 32);
            float q = v + fcr[u][r] * p;
            dnm[u][r] = fmaxf(fabsf(q), 1.0f);
        }

#pragma unroll
    for (int u = 0; u < 2; u++)
#pragma unroll
        for (int dt = 0; dt < 8; dt++)
#pragma unroll
            for (int r = 0; r < 8; r++) {
                int row = m0 + u * 16 + r + 8 * half;
                int col = dt * 16 + mn;
                float ov = opre[(size_t)(b * S_ + row) * D_ + col];
                float sg = 1.f / (1.f + expf(-ov));
                outh[(size_t)(b * S_ + row) * D_ + col] = sg * acc[u][dt][r] / dnm[u][r];
            }
}

// ---------------------------------------------------------------------------
// 7) C_last = f_cum[S-1]*C_prev + V^T @ (i*K): 32x32 register tile per wave.
// ---------------------------------------------------------------------------
__global__ void clast_kernel(const unsigned short* __restrict__ vT,
                             const unsigned short* __restrict__ kkT,
                             const float* __restrict__ cprev,
                             const float* __restrict__ fc,
                             float* __restrict__ outC)
{
    int b = blockIdx.z;
    int d0 = blockIdx.x * 32, e0 = blockIdx.y * 32;
    int lane = threadIdx.x & 31;
    int half = lane >> 4, mn = lane & 15;
    const unsigned short* a0p = vT  + (size_t)b * D_ * S_ + (size_t)(d0 + mn) * S_ + half * 8;
    const unsigned short* a1p = a0p + (size_t)16 * S_;
    const unsigned short* b0p = kkT + (size_t)b * D_ * S_ + (size_t)(e0 + mn) * S_ + half * 8;
    const unsigned short* b1p = b0p + (size_t)16 * S_;
    v8f acc00 = {}, acc01 = {}, acc10 = {}, acc11 = {};
    for (int kb = 0; kb < S_; kb += 32) {
        v16bf a0 = load_frag(a0p + kb);
        v16bf a1 = load_frag(a1p + kb);
        v16bf b0 = load_frag(b0p + kb);
        v16bf b1 = load_frag(b1p + kb);
        acc00 = wmma_bf16(a0, b0, acc00);
        acc01 = wmma_bf16(a0, b1, acc01);
        acc10 = wmma_bf16(a1, b0, acc10);
        acc11 = wmma_bf16(a1, b1, acc11);
    }
    float fcl = fc[b * S_ + S_ - 1];
    const float* cp = cprev + (size_t)b * D_ * D_;
    float* oc = outC + (size_t)b * D_ * D_;
#pragma unroll
    for (int r = 0; r < 8; r++) {
        int row0 = d0 + r + 8 * half, row1 = row0 + 16;
        int col0 = e0 + mn, col1 = col0 + 16;
        oc[(size_t)row0 * D_ + col0] = acc00[r] + fcl * cp[(size_t)row0 * D_ + col0];
        oc[(size_t)row0 * D_ + col1] = acc01[r] + fcl * cp[(size_t)row0 * D_ + col1];
        oc[(size_t)row1 * D_ + col0] = acc10[r] + fcl * cp[(size_t)row1 * D_ + col0];
        oc[(size_t)row1 * D_ + col1] = acc11[r] + fcl * cp[(size_t)row1 * D_ + col1];
    }
}

// ---------------------------------------------------------------------------
// 8) n_last[b,d] = sum_t i_t k[t,d] + f_cum[S-1]*n_prev
// ---------------------------------------------------------------------------
__global__ void nlast_kernel(const float* __restrict__ kf, const float* __restrict__ ia,
                             const float* __restrict__ fc, const float* __restrict__ nprev,
                             float* __restrict__ outN)
{
    int b = blockIdx.x, d = threadIdx.x;
    float s = 0.f;
    for (int t = 0; t < S_; t++)
        s += ia[b * S_ + t] * kf[((size_t)(b * S_ + t)) * D_ + d];
    outN[b * D_ + d] = s + fc[b * S_ + S_ - 1] * nprev[b * D_ + d];
}

// ---------------------------------------------------------------------------
extern "C" void kernel_launch(void* const* d_in, const int* in_sizes, int n_in,
                              void* d_out, int out_size, void* d_ws, size_t ws_size,
                              hipStream_t stream)
{
    const float* x      = (const float*)d_in[0];
    const float* C_prev = (const float*)d_in[1];
    const float* n_prev = (const float*)d_in[2];
    const float* m_prev = (const float*)d_in[3];
    const float* Wq = (const float*)d_in[4];  const float* bq = (const float*)d_in[5];
    const float* Wk = (const float*)d_in[6];  const float* bk = (const float*)d_in[7];
    const float* Wv = (const float*)d_in[8];  const float* bv = (const float*)d_in[9];
    const float* Wi = (const float*)d_in[10]; const float* bi = (const float*)d_in[11];
    const float* Wf = (const float*)d_in[12]; const float* bfv = (const float*)d_in[13];
    const float* Wo = (const float*)d_in[14]; const float* bo = (const float*)d_in[15];
    const float* ln_g = (const float*)d_in[16]; const float* ln_b = (const float*)d_in[17];

    const int BS = B_ * S_;                      // 4096 rows
    char* ws = (char*)d_ws;
    size_t off = 0;
    auto take = [&](size_t bytes) -> void* {
        void* p = ws + off; off = (off + bytes + 255) & ~(size_t)255; return p;
    };
    unsigned short* xn  = (unsigned short*)take((size_t)BS * IN_ * 2);
    unsigned short* wqb = (unsigned short*)take((size_t)D_ * IN_ * 2);
    unsigned short* wkb = (unsigned short*)take((size_t)D_ * IN_ * 2);
    unsigned short* wvb = (unsigned short*)take((size_t)D_ * IN_ * 2);
    unsigned short* wob = (unsigned short*)take((size_t)D_ * IN_ * 2);
    unsigned short* cpb = (unsigned short*)take((size_t)B_ * D_ * D_ * 2);
    float* qf = (float*)take((size_t)BS * D_ * 4);
    float* kf = (float*)take((size_t)BS * D_ * 4);
    float* vf = (float*)take((size_t)BS * D_ * 4);
    float* of = (float*)take((size_t)BS * D_ * 4);
    unsigned short* qbf  = (unsigned short*)take((size_t)BS * D_ * 2);
    unsigned short* kkbf = (unsigned short*)take((size_t)BS * D_ * 2);
    unsigned short* kkT  = (unsigned short*)take((size_t)B_ * D_ * S_ * 2);
    unsigned short* vT   = (unsigned short*)take((size_t)B_ * D_ * S_ * 2);
    float* itld = (float*)take((size_t)BS * 4);
    float* ftld = (float*)take((size_t)BS * 4);
    float* ia   = (float*)take((size_t)BS * 4);
    float* fcum = (float*)take((size_t)BS * 4);
    (void)ws_size; (void)in_sizes; (void)n_in; (void)out_size;

    float* out_h = (float*)d_out;                        // (B,S,D)
    float* out_C = out_h + (size_t)BS * D_;              // (B,D,D)
    float* out_n = out_C + (size_t)B_ * D_ * D_;         // (B,D)
    float* out_m = out_n + (size_t)B_ * D_;              // (B,1)

    // 1) LayerNorm + gate dots
    ln_kernel<<<BS, 256, 0, stream>>>(x, ln_g, ln_b, Wi, bi, Wf, bfv, xn, itld, ftld);

    // 2) bf16 conversions
    int wn = D_ * IN_;
    cvt_bf<<<(wn + 255) / 256, 256, 0, stream>>>(Wq, wqb, wn);
    cvt_bf<<<(wn + 255) / 256, 256, 0, stream>>>(Wk, wkb, wn);
    cvt_bf<<<(wn + 255) / 256, 256, 0, stream>>>(Wv, wvb, wn);
    cvt_bf<<<(wn + 255) / 256, 256, 0, stream>>>(Wo, wob, wn);
    int cn = B_ * D_ * D_;
    cvt_bf<<<(cn + 255) / 256, 256, 0, stream>>>(C_prev, cpb, cn);

    // 3) gate scan (writes m_last)
    scan_kernel<<<B_, 32, 0, stream>>>(itld, ftld, m_prev, ia, fcum, out_m);

    // 4) projections: (4096x512) @ (512x128), 32x32 tiles
    dim3 ggrid(BS / 32, D_ / 32);
    const float kscale = 0.08838834764831845f;           // 1/sqrt(128)
    gemm32<<<ggrid, 32, 0, stream>>>(xn, wqb, bq, qf, BS, D_, IN_, 1.0f);
    gemm32<<<ggrid, 32, 0, stream>>>(xn, wkb, bk, kf, BS, D_, IN_, kscale);
    gemm32<<<ggrid, 32, 0, stream>>>(xn, wvb, bv, vf, BS, D_, IN_, 1.0f);
    gemm32<<<ggrid, 32, 0, stream>>>(xn, wob, bo, of, BS, D_, IN_, 1.0f);

    // 5) pack bf16 operands (fold i into k, build transposed layouts)
    pack_kernel<<<BS, D_, 0, stream>>>(qf, kf, vf, ia, qbf, kkbf, kkT, vT);

    // 6) causal linear attention -> h_t (32-query tiles)
    attn_kernel<<<dim3(S_ / 32, B_), 32, 0, stream>>>(qbf, kkbf, vT, cpb, of, fcum,
                                                      n_prev, out_h);

    // 7) C_last, 8) n_last
    clast_kernel<<<dim3(D_ / 32, D_ / 32, B_), 32, 0, stream>>>(vT, kkT, C_prev, fcum, out_C);
    nlast_kernel<<<B_, D_, 0, stream>>>(kf, ia, fcum, n_prev, out_n);
}
